// Attention_85882166051391
// MI455X (gfx1250) — compile-verified, matrix operand order checked by reading, not run on hardware
//
#include <hip/hip_runtime.h>

// ---------------------------------------------------------------------------
// Problem constants (from reference): B=8, S=1024, H=1024, NH=4, D=256
// ---------------------------------------------------------------------------
#define B_  8
#define S_  1024
#define H_  1024
#define NH_ 4
#define D_  256
#define M_TOTAL (B_ * S_)          // 8192
#define NEGVAL  (-4294967295.0f)   // -2^32 + 1

typedef __bf16 bf16_t;
typedef __attribute__((ext_vector_type(16))) __bf16 v16bf;
typedef __attribute__((ext_vector_type(8)))  float  v8f;
typedef __attribute__((ext_vector_type(4)))  int    v4i;

#define AS1 __attribute__((address_space(1)))
#define AS3 __attribute__((address_space(3)))

#if defined(__has_builtin)
#if __has_builtin(__builtin_amdgcn_global_load_async_to_lds_b128)
#define USE_ASYNC_LDS 1
#endif
#endif

__device__ __forceinline__ void async_wait0() {
#if defined(USE_ASYNC_LDS)
#if __has_builtin(__builtin_amdgcn_s_wait_asynccnt)
  __builtin_amdgcn_s_wait_asynccnt(0);
#else
  asm volatile("s_wait_asynccnt 0x0" ::: "memory");
#endif
#endif
}

// stage one 16-byte chunk from global into LDS (async DMA when available)
__device__ __forceinline__ void stage16(const bf16_t* gsrc, bf16_t* ldst) {
#if defined(USE_ASYNC_LDS)
  __builtin_amdgcn_global_load_async_to_lds_b128((AS1 v4i*)gsrc, (AS3 v4i*)ldst,
                                                 0, 0);
#else
  *(uint4*)ldst = *(const uint4*)gsrc;
#endif
}

// ---------------------------------------------------------------------------
// Kernel 1a: plain f32 -> bf16 cast
// ---------------------------------------------------------------------------
__global__ void cast_bf16_kernel(const float* __restrict__ x,
                                 bf16_t* __restrict__ y, int n) {
  int i = blockIdx.x * blockDim.x + threadIdx.x;
  if (i < n) y[i] = (bf16_t)x[i];
}

// ---------------------------------------------------------------------------
// Kernel 1b: f32 -> bf16 transposed cast: out(N,K) = in(K,N)^T
// ---------------------------------------------------------------------------
__global__ void cast_transpose_kernel(const float* __restrict__ in,
                                      bf16_t* __restrict__ out, int K, int N) {
  int idx = blockIdx.x * blockDim.x + threadIdx.x;
  if (idx < K * N) {
    int n = idx / K, k = idx % K;
    out[idx] = (bf16_t)in[(size_t)k * N + n];
  }
}

// ---------------------------------------------------------------------------
// Kernel 2: bf16 WMMA GEMM with pre-transposed B, double-buffered async LDS.
//   C(M,N) = A(M,K) @ B, where Bt holds B^T (N rows of K contiguous values).
//   Block = 256 threads (8 wave32 waves). Block tile: 128(M) x 128(N).
//   Each wave owns a 16x128 strip: 8 accumulators, 8 WMMAs per K-step for
//   one A-fragment. B tile (128x32 = 8KB) is staged through LDS with
//   GLOBAL_LOAD_ASYNC_TO_LDS_B128 into the alternate buffer while the
//   current one feeds the WMMAs; one s_wait_asynccnt + barrier per K-step.
// ---------------------------------------------------------------------------
__global__ void gemm_bf16_nt_kernel(const bf16_t* __restrict__ A, int lda,
                                    const bf16_t* __restrict__ Bt, int ldb,
                                    float* Cf, bf16_t* Cbf, int ldc,
                                    const float* __restrict__ bias,
                                    int M, int N, int K,
                                    int accFlag, int transC) {
  __shared__ __align__(16) bf16_t Bs[2][128 * 32];   // [buf][n_local][k_local]

  const int m0   = blockIdx.y * 128 + (threadIdx.x >> 5) * 16;
  const int n0   = blockIdx.x * 128;
  const int tid  = threadIdx.x;
  const int lane = tid & 31;
  const int r16  = lane & 15;
  const int half = lane >> 4;

  // cooperative copy: 128 rows x 32 k = 8KB; thread t moves 32B of row t/2
  const int crow = tid >> 1;               // 0..127 (n_local)
  const int cko  = (tid & 1) * 16;         // 0 or 16 (k_local)

  const bf16_t* arow  = A + (size_t)(m0 + r16) * lda;
  const bf16_t* bbase = Bt + (size_t)(n0 + crow) * ldb + cko;
  bf16_t* l0 = &Bs[0][crow * 32 + cko];
  bf16_t* l1 = &Bs[1][crow * 32 + cko];

  v8f acc[8];
#pragma unroll
  for (int t = 0; t < 8; ++t) acc[t] = (v8f){};

  const int ntiles = K / 32;

  // prologue: stage tile 0 into buffer 0
  stage16(bbase, l0);
  stage16(bbase + 8, l0 + 8);
  async_wait0();
  __syncthreads();

  for (int i = 0; i < ntiles; ++i) {
    const int k0 = i * 32;
    bf16_t* cur = (i & 1) ? &Bs[1][0] : &Bs[0][0];
    // stage next tile into the alternate buffer; DMA overlaps the WMMAs below
    if (i + 1 < ntiles) {
      const bf16_t* gnext = bbase + (size_t)(i + 1) * 32;
      bf16_t* lnext = (i & 1) ? l0 : l1;
      stage16(gnext, lnext);
      stage16(gnext + 8, lnext + 8);
      __builtin_prefetch(arow + k0 + 32, 0, 1);   // global_prefetch_b8
    }

    v16bf af = *(const v16bf*)(arow + k0 + half * 16);
#pragma unroll
    for (int t = 0; t < 8; ++t) {
      v16bf bfrag = *(const v16bf*)&cur[(t * 16 + r16) * 32 + half * 16];
      acc[t] = __builtin_amdgcn_wmma_f32_16x16x32_bf16(false, af, false, bfrag,
                                                       (short)0, acc[t],
                                                       false, false);
    }

    async_wait0();      // next tile resident in LDS
    __syncthreads();    // all waves done reading `cur` before it is re-staged
  }

  // epilogue: VGPR j of tile t -> row (half*8 + j), col = n0 + t*16 + lane%16
#pragma unroll
  for (int t = 0; t < 8; ++t) {
    const int col = n0 + t * 16 + r16;
#pragma unroll
    for (int j = 0; j < 8; ++j) {
      int r = m0 + half * 8 + j;
      float v = acc[t][j];
      if (bias) v += bias[col];
      if (Cf) {
        size_t o = (size_t)r * ldc + col;
        if (accFlag) v += Cf[o];
        Cf[o] = v;
      }
      if (Cbf) {
        size_t o = transC ? ((size_t)col * ldc + r) : ((size_t)r * ldc + col);
        Cbf[o] = (bf16_t)v;
      }
    }
  }
}

// ---------------------------------------------------------------------------
// Kernel 3: fused scores + softmax for one 16-row q stripe of one (h,b).
//   grid = (S/16, NH*B), block = 128 (4 waves).
// ---------------------------------------------------------------------------
__global__ void scores_softmax_kernel(const bf16_t* __restrict__ Qbf,
                                      const bf16_t* __restrict__ Kbf,
                                      const unsigned char* __restrict__ mask,
                                      const float* __restrict__ g_factor,
                                      const float* __restrict__ query_mask,
                                      float* __restrict__ attn,
                                      bf16_t* __restrict__ attn_bf) {
  __shared__ float sc[16 * S_];   // 64 KB stripe

  const int q0   = blockIdx.x * 16;
  const int hb   = blockIdx.y;        // h*B + b
  const int b    = hb % B_;
  const int h    = hb / B_;
  const int tid  = threadIdx.x;
  const int lane = tid & 31;
  const int wave = tid >> 5;
  const int r16  = lane & 15;
  const int half = lane >> 4;

  // ---- phase 1: QK^T tiles into LDS ----
  v16bf aq[8];
  const bf16_t* qbase = Qbf + (size_t)(b * S_ + q0 + r16) * H_ + h * D_;
#pragma unroll
  for (int kd = 0; kd < 8; ++kd)
    aq[kd] = *(const v16bf*)(qbase + kd * 32 + half * 16);

  const float gf    = g_factor[0];
  const float scale = 0.0625f;        // 1/sqrt(256)

  for (int kt = wave; kt < S_ / 16; kt += 4) {
    const int k0 = kt * 16;
    const bf16_t* kbase = Kbf + (size_t)(b * S_ + k0 + r16) * H_ + h * D_;
    v8f c = {};
#pragma unroll
    for (int kd = 0; kd < 8; ++kd) {
      v16bf bk = *(const v16bf*)(kbase + kd * 32 + half * 16); // K^T fragment
      c = __builtin_amdgcn_wmma_f32_16x16x32_bf16(false, aq[kd], false, bk,
                                                  (short)0, c, false, false);
    }
    const int kcol = k0 + r16;
#pragma unroll
    for (int j = 0; j < 8; ++j) {
      int ql = half * 8 + j;
      int q  = q0 + ql;
      float dqk = (float)(q - kcol);
      float v = c[j] * scale - dqk * dqk / gf;
      if (mask[((size_t)b * S_ + q) * S_ + kcol]) v = NEGVAL;
      sc[ql * S_ + kcol] = v;
    }
  }
  __syncthreads();

  // ---- phase 2: row softmax from LDS ----
  const int row = tid >> 3;           // 0..15
  const int sub = tid & 7;            // 8 lanes cooperate per row
  const int q   = q0 + row;
  float* rowp = &sc[row * S_];

  float m = -3.4e38f;
  for (int c = sub; c < S_; c += 8) m = fmaxf(m, rowp[c]);
#pragma unroll
  for (int o = 1; o < 8; o <<= 1) m = fmaxf(m, __shfl_xor(m, o, 32));

  float sum = 0.0f;
  for (int c = sub; c < S_; c += 8) {
    float e = __expf(rowp[c] - m);
    rowp[c] = e;
    sum += e;
  }
#pragma unroll
  for (int o = 1; o < 8; o <<= 1) sum += __shfl_xor(sum, o, 32);

  const float inv = query_mask[b * S_ + q] / sum;
  float* orow  = attn    + (((size_t)hb) * S_ + q) * S_;
  bf16_t* brow = attn_bf + (((size_t)hb) * S_ + q) * S_;
  for (int c = sub; c < S_; c += 8) {
    float v = rowp[c] * inv;
    orow[c] = v;
    brow[c] = (bf16_t)v;
  }
}

// ---------------------------------------------------------------------------
// Kernel 4: residual add + LayerNorm
// ---------------------------------------------------------------------------
__global__ void ln_kernel(const float* __restrict__ F,
                          const float* __restrict__ dec,
                          const float* __restrict__ gamma,
                          const float* __restrict__ beta,
                          float* __restrict__ out) {
  const int row = blockIdx.x;
  const float* f = F   + (size_t)row * H_;
  const float* d = dec + (size_t)row * H_;
  float* o       = out + (size_t)row * H_;
  const int tid = threadIdx.x;
  __shared__ float red[256];

  float s = 0.0f;
  for (int i = tid; i < H_; i += 256) s += f[i] + d[i];
  red[tid] = s;
  __syncthreads();
  for (int st = 128; st > 0; st >>= 1) {
    if (tid < st) red[tid] += red[tid + st];
    __syncthreads();
  }
  const float mu = red[0] / (float)H_;
  __syncthreads();

  float vs = 0.0f;
  for (int i = tid; i < H_; i += 256) {
    float x = f[i] + d[i] - mu;
    vs += x * x;
  }
  red[tid] = vs;
  __syncthreads();
  for (int st = 128; st > 0; st >>= 1) {
    if (tid < st) red[tid] += red[tid + st];
    __syncthreads();
  }
  const float inv = rsqrtf(red[0] / (float)H_ + 1e-5f);
  __syncthreads();

  for (int i = tid; i < H_; i += 256) {
    float x = (f[i] + d[i] - mu) * inv;
    o[i] = x * gamma[i] + beta[i];
  }
}

// ---------------------------------------------------------------------------
// Host orchestration
// ---------------------------------------------------------------------------
static inline size_t align256(size_t x) { return (x + 255) & ~(size_t)255; }

extern "C" void kernel_launch(void* const* d_in, const int* in_sizes, int n_in,
                              void* d_out, int out_size, void* d_ws, size_t ws_size,
                              hipStream_t stream) {
  (void)in_sizes; (void)n_in; (void)out_size; (void)ws_size;

  const float* memory   = (const float*)d_in[0];
  const float* dec      = (const float*)d_in[1];
  const unsigned char* mask = (const unsigned char*)d_in[2];
  const float* qmask    = (const float*)d_in[3];
  const float* Wk       = (const float*)d_in[4];
  const float* Wv       = (const float*)d_in[5];
  const float* Wq       = (const float*)d_in[6];
  const float* Wf       = (const float*)d_in[7];
  const float* bfv      = (const float*)d_in[8];
  const float* gamma    = (const float*)d_in[9];
  const float* beta     = (const float*)d_in[10];
  const float* gfac     = (const float*)d_in[11];

  float* out_f  = (float*)d_out;                                   // (B,S,H)
  float* attn_f = out_f + (size_t)M_TOTAL * H_;                    // (NH*B,S,S)

  // workspace carve-up
  char* w = (char*)d_ws;
  size_t off = 0;
  auto take = [&](size_t bytes) { void* p = w + off; off = align256(off + bytes); return p; };
  const size_t NE = (size_t)M_TOTAL * H_;          // 8M activations
  bf16_t* mem_bf  = (bf16_t*)take(NE * 2);
  bf16_t* dec_bf  = (bf16_t*)take(NE * 2);
  bf16_t* WkT     = (bf16_t*)take((size_t)H_ * H_ * 2);       // (N=H, K=H)
  bf16_t* WvT     = (bf16_t*)take((size_t)H_ * H_ * 2);
  bf16_t* WqT     = (bf16_t*)take((size_t)H_ * H_ * 2);
  bf16_t* WfT     = (bf16_t*)take((size_t)H_ * 2 * H_ * 2);   // (1024 x 2048)
  bf16_t* Kbf     = (bf16_t*)take(NE * 2);                    // (8192 x 1024)
  bf16_t* Vt      = (bf16_t*)take(NE * 2);                    // (1024 x 8192)
  bf16_t* Qbf     = (bf16_t*)take(NE * 2);
  bf16_t* attn_bf = (bf16_t*)take((size_t)NH_ * B_ * S_ * S_ * 2);
  bf16_t* res_bf  = (bf16_t*)take(NE * 2);
  float*  Facc    = (float*) take(NE * 4);

  // 1) casts / transposed casts to bf16
  auto cast = [&](const float* src, bf16_t* dst, size_t n) {
    cast_bf16_kernel<<<dim3((unsigned)((n + 255) / 256)), 256, 0, stream>>>(src, dst, (int)n);
  };
  auto castT = [&](const float* src, bf16_t* dst, int K, int N) {
    size_t n = (size_t)K * N;
    cast_transpose_kernel<<<dim3((unsigned)((n + 255) / 256)), 256, 0, stream>>>(src, dst, K, N);
  };
  cast(memory, mem_bf, NE);
  cast(dec,    dec_bf, NE);
  castT(Wk, WkT, H_, H_);
  castT(Wv, WvT, H_, H_);
  castT(Wq, WqT, H_, H_);
  castT(Wf, WfT, 2 * H_, H_);

  // 2) projections (block tile 128x128)
  dim3 gproj(H_ / 128, M_TOTAL / 128);   // (8, 64)
  gemm_bf16_nt_kernel<<<gproj, 256, 0, stream>>>(mem_bf, H_, WkT, H_,
                                                 nullptr, Kbf, H_, nullptr,
                                                 M_TOTAL, H_, H_, 0, 0);
  gemm_bf16_nt_kernel<<<gproj, 256, 0, stream>>>(mem_bf, H_, WvT, H_,
                                                 nullptr, Vt, M_TOTAL, nullptr,
                                                 M_TOTAL, H_, H_, 0, /*transC=*/1);
  gemm_bf16_nt_kernel<<<gproj, 256, 0, stream>>>(dec_bf, H_, WqT, H_,
                                                 nullptr, Qbf, H_, nullptr,
                                                 M_TOTAL, H_, H_, 0, 0);

  // 3) fused scores + softmax -> attn (f32 in d_out) + attn_bf (workspace)
  scores_softmax_kernel<<<dim3(S_ / 16, NH_ * B_), 128, 0, stream>>>(
      Qbf, Kbf, mask, gfac, qmask, attn_f, attn_bf);

  // 4) res = attn @ V per (h,b) slice (B supplied transposed via Vt)
  for (int hb = 0; hb < NH_ * B_; ++hb) {
    int b = hb % B_, h = hb / B_;
    const bf16_t* Aslice  = attn_bf + (size_t)hb * S_ * S_;
    const bf16_t* BtSlice = Vt + (size_t)(h * D_) * M_TOTAL + b * S_;
    bf16_t*       Cslice  = res_bf + (size_t)b * S_ * H_ + h * D_;
    dim3 gav(D_ / 128, S_ / 128);  // (2, 8)
    gemm_bf16_nt_kernel<<<gav, 256, 0, stream>>>(Aslice, S_, BtSlice, M_TOTAL,
                                                 nullptr, Cslice, H_, nullptr,
                                                 S_, D_, S_, 0, 0);
  }

  // 5) F = dec@Wf_top + bf ; F += res@Wf_bot  (WfT rows hold K=2048 contiguous)
  gemm_bf16_nt_kernel<<<gproj, 256, 0, stream>>>(dec_bf, H_, WfT, 2 * H_,
                                                 Facc, nullptr, H_, bfv,
                                                 M_TOTAL, H_, H_, 0, 0);
  gemm_bf16_nt_kernel<<<gproj, 256, 0, stream>>>(res_bf, H_, WfT + H_, 2 * H_,
                                                 Facc, nullptr, H_, nullptr,
                                                 M_TOTAL, H_, H_, 1, 0);

  // 6) residual + LayerNorm -> first 8M floats of d_out
  ln_kernel<<<dim3(M_TOTAL), 256, 0, stream>>>(Facc, dec, gamma, beta, out_f);
}
